// VectorQuantizer_4793183502752
// MI455X (gfx1250) — matrix operand order repaired and checked by reading, not compiled, not run on hardware
//
#include <hip/hip_runtime.h>

typedef unsigned short u16;
typedef __attribute__((ext_vector_type(16))) __bf16 v16bf;
typedef __attribute__((ext_vector_type(8)))  float  v8f;

#define Bsz   16
#define Cdim  256
#define Ldim  4096
#define Kcb   4096
#define NTILE 128
#define THREADS 256

__device__ __forceinline__ u16 f2bf(float f){
  unsigned u = __float_as_uint(f);
  u += 0x7FFFu + ((u >> 16) & 1u);      // round-to-nearest-even (inputs are finite)
  return (u16)(u >> 16);
}
__device__ __forceinline__ float bf2f(u16 h){
  return __uint_as_float(((unsigned)h) << 16);
}

// ---------------------------------------------------------------------------
// Kernel 1: pack codebook into WMMA A-fragment layout (bf16 hi + lo) + ||c||^2
// A order used here (not AMD's table order): lane m    holds K = {0..15}
//                             lane m+16 holds K = {16..31}
// pack layout: [kt][cc][lane][slot] halves, 8KB per kt-block.
// ---------------------------------------------------------------------------
__global__ __launch_bounds__(256) void vq_pack(const float* __restrict__ cb,
                                               u16* __restrict__ pHi,
                                               u16* __restrict__ pLo,
                                               float* __restrict__ csq){
  __shared__ float cbs[16 * Cdim];
  const int t  = threadIdx.x;
  const int kt = blockIdx.x;                       // 16 codewords per block
  const float* src = cb + (size_t)kt * 16 * Cdim;
  for (int e = t; e < 16 * Cdim; e += THREADS) cbs[e] = src[e];
  __syncthreads();
  if (t < 16){
    float s = 0.f;
    for (int c = 0; c < Cdim; ++c){ float v = cbs[t * Cdim + c]; s += v * v; }
    csq[kt * 16 + t] = s;
  }
  const int cc   = t >> 5;                         // 0..7  (K chunk of 32)
  const int lane = t & 31;
  const int m    = lane & 15;
  alignas(16) u16 hi[16], lo[16];
#pragma unroll
  for (int s = 0; s < 16; ++s){
    const int kl = (lane < 16) ? s
                               : (16 + s);
    const float v = cbs[m * Cdim + cc * 32 + kl];
    const u16 h = f2bf(v);
    hi[s] = h;
    lo[s] = f2bf(v - bf2f(h));
  }
  const size_t base = ((size_t)kt << 12) + cc * 512 + lane * 16;
  *(uint4*)(pHi + base)     = *(const uint4*)&hi[0];
  *(uint4*)(pHi + base + 8) = *(const uint4*)&hi[8];
  *(uint4*)(pLo + base)     = *(const uint4*)&lo[0];
  *(uint4*)(pLo + base + 8) = *(const uint4*)&lo[8];
}

// ---------------------------------------------------------------------------
// Kernel 2: fused GEMM + argmin.
// Each block: 128 x-rows (one batch b, l-tile of 128); 8 waves, wave w owns
// rows [w*16, w*16+16).  B-fragments (x) are staged through one 64KB LDS
// buffer (hi pass, then lo pass) into persistent VGPRs; the K loop then only
// streams packed A-fragments (codebook) from L2/WGP$.
// 3 bf16 WMMAs per c-chunk (hi*hi + hi*lo + lo*hi) into 3 f32 accumulators.
// B order used here (matches A; not AMD's table order): lane n holds col n K=0..15, lane n+16 K=16..31.
// ---------------------------------------------------------------------------
__global__ __launch_bounds__(256) void vq_main(const float* __restrict__ x,
                                               const u16* __restrict__ pHi,
                                               const u16* __restrict__ pLo,
                                               const float* __restrict__ csq,
                                               int*   __restrict__ idxW,
                                               float* __restrict__ idxF){
  __shared__ u16 xb[8 * 8 * 32 * 16];   // [cc][nt][lane][slot]  64 KB (reused)
  const int t  = threadIdx.x;
  const int b  = blockIdx.x >> 5;
  const int l0 = (blockIdx.x & 31) << 7;
  const float* xg = x + ((size_t)b * Cdim) * Ldim + l0;

  const int w    = t >> 5;
  const int lane = t & 31;
  const int hl   = lane >> 4;
  const int n    = lane & 15;
  const u16* bfrag = &xb[(w << 9) + (lane << 4)];   // cc stride = 4096 halves

  v16bf Bh[8], Bl[8];

  // Pass 1: bf16-hi tile -> LDS -> registers (coalesced global reads along l).
  for (int e = t; e < NTILE * Cdim; e += THREADS){
    const int c = e >> 7;
    const int i = e & 127;
    const float v = xg[(size_t)c * Ldim + i];
    const int a = ((((c >> 5) << 3) + (i >> 4)) << 9) |
                  (((i & 15) | (((c >> 4) & 1) << 4)) << 4) | (c & 15);
    xb[a] = f2bf(v);
  }
  __syncthreads();
#pragma unroll
  for (int cc = 0; cc < 8; ++cc) Bh[cc] = *(const v16bf*)(bfrag + cc * 4096);
  __syncthreads();

  // Pass 2: bf16-lo residual tile -> LDS -> registers.
  for (int e = t; e < NTILE * Cdim; e += THREADS){
    const int c = e >> 7;
    const int i = e & 127;
    const float v = xg[(size_t)c * Ldim + i];
    const u16 h = f2bf(v);
    const int a = ((((c >> 5) << 3) + (i >> 4)) << 9) |
                  (((i & 15) | (((c >> 4) & 1) << 4)) << 4) | (c & 15);
    xb[a] = f2bf(v - bf2f(h));
  }
  __syncthreads();
#pragma unroll
  for (int cc = 0; cc < 8; ++cc) Bl[cc] = *(const v16bf*)(bfrag + cc * 4096);

  float best = 3.4028235e38f;
  int   bidx = 0;

  for (int kt = 0; kt < Kcb / 16; ++kt){
    const u16* ah = pHi + ((size_t)kt << 12);
    const u16* al = pLo + ((size_t)kt << 12);
    // Next-block prefetch; one-past-the-end lands inside our workspace and
    // speculative prefetches are dropped on translation failure anyway.
    __builtin_prefetch(pHi + ((size_t)(kt + 1) << 12) + (lane << 8), 0, 1);
    __builtin_prefetch(pLo + ((size_t)(kt + 1) << 12) + (lane << 8), 0, 1);

    v8f aHH = {}, aHL = {}, aLH = {};
#pragma unroll
    for (int cc = 0; cc < 8; ++cc){
      const v16bf Ah = *(const v16bf*)(ah + cc * 512 + (lane << 4));
      const v16bf Al = *(const v16bf*)(al + cc * 512 + (lane << 4));
      aHH = __builtin_amdgcn_wmma_f32_16x16x32_bf16(false, Ah, false, Bh[cc], (short)0, aHH, false, false);
      aHL = __builtin_amdgcn_wmma_f32_16x16x32_bf16(false, Ah, false, Bl[cc], (short)0, aHL, false, false);
      aLH = __builtin_amdgcn_wmma_f32_16x16x32_bf16(false, Al, false, Bh[cc], (short)0, aLH, false, false);
    }
    // scores: ||c_k||^2 - 2*dot ; D tile: acc[r] = (codeword r+8*hl, row n)
    const float* cs = csq + (kt << 4) + (hl << 3);
    const float4 c0 = *(const float4*)cs;
    const float4 c1 = *(const float4*)(cs + 4);
    const float cq[8] = {c0.x, c0.y, c0.z, c0.w, c1.x, c1.y, c1.z, c1.w};
#pragma unroll
    for (int r = 0; r < 8; ++r){
      const float dot = aHH[r] + aHL[r] + aLH[r];
      const float sc  = cq[r] - 2.0f * dot;
      const int   k   = (kt << 4) + (hl << 3) + r;
      if (sc < best){ best = sc; bidx = k; }
    }
  }

  // merge lane-halves (row n lives in lanes n and n+16); ties -> smaller index
  const float ov = __shfl_xor(best, 16, 32);
  const int   oi = __shfl_xor(bidx, 16, 32);
  if (ov < best || (ov == best && oi < bidx)){ best = ov; bidx = oi; }

  if (hl == 0){
    const size_t gi = (size_t)b * Ldim + l0 + (w << 4) + n;
    idxW[gi] = bidx;
    idxF[gi] = (float)bidx;     // indices output (value-cast to f32)
  }
}

// ---------------------------------------------------------------------------
// Kernel 3: z_q gather + verbatim x copy (coalesced along l, NT stores).
// ---------------------------------------------------------------------------
__global__ __launch_bounds__(256) void vq_out(const float* __restrict__ x,
                                              const float* __restrict__ cb,
                                              const int* __restrict__ idxW,
                                              float* __restrict__ zq,
                                              float* __restrict__ xcopy){
  __shared__ int il[NTILE];
  const int t  = threadIdx.x;
  const int b  = blockIdx.x >> 5;
  const int l0 = (blockIdx.x & 31) << 7;
  if (t < NTILE) il[t] = idxW[(size_t)b * Ldim + l0 + t];
  __syncthreads();
  const float* xg = x     + ((size_t)b * Cdim) * Ldim + l0;
  float*       zg = zq    + ((size_t)b * Cdim) * Ldim + l0;
  float*       cg = xcopy + ((size_t)b * Cdim) * Ldim + l0;
  for (int e = t; e < NTILE * Cdim; e += THREADS){
    const int c = e >> 7;
    const int i = e & 127;
    const size_t off = (size_t)c * Ldim + i;
    const float z = cb[(size_t)il[i] * Cdim + c];
    __builtin_nontemporal_store(z, zg + off);
    __builtin_nontemporal_store(__builtin_nontemporal_load(xg + off), cg + off);
  }
}

// ---------------------------------------------------------------------------
extern "C" void kernel_launch(void* const* d_in, const int* in_sizes, int n_in,
                              void* d_out, int out_size, void* d_ws, size_t ws_size,
                              hipStream_t stream) {
  const float* x  = (const float*)d_in[0];   // [16,256,4096] f32
  const float* cb = (const float*)d_in[1];   // [4096,256] f32

  float* zq    = (float*)d_out;                               // [16,256,4096]
  float* xcopy = zq    + (size_t)Bsz * Cdim * Ldim;           // [16,256,4096]
  float* idxF  = xcopy + (size_t)Bsz * Cdim * Ldim;           // [16,4096]

  char* ws = (char*)d_ws;
  float* csq = (float*)ws;                                    // 16 KB
  u16*   pHi = (u16*)(ws + 16384);                            // 2 MB
  u16*   pLo = (u16*)(ws + 16384 + (2u << 20));               // 2 MB
  int*   idxW = (int*)(ws + 16384 + (4u << 20));              // 256 KB

  hipLaunchKernelGGL(vq_pack, dim3(Kcb / 16), dim3(THREADS), 0, stream, cb, pHi, pLo, csq);
  hipLaunchKernelGGL(vq_main, dim3((Bsz * Ldim) / NTILE), dim3(THREADS), 0, stream,
                     x, pHi, pLo, csq, idxW, idxF);
  hipLaunchKernelGGL(vq_out,  dim3((Bsz * Ldim) / NTILE), dim3(THREADS), 0, stream,
                     x, cb, idxW, zq, xcopy);
}
